// LogicalGNN_50629074485936
// MI455X (gfx1250) — matrix operand order repaired and checked by reading, not compiled, not run on hardware
//
#include <hip/hip_runtime.h>

// ---------------------------------------------------------------------------
// LogicalGNN for MI455X (gfx1250): bf16 WMMA GEMMs + f32 scatter softmax.
// GEMM: one wave owns a 32x64 C macro-tile = 2x4 grid of 16x16 WMMA tiles.
// Per K-step (32): 2 A-fragment loads + 4 B-fragment loads -> 8 v_wmma,
// ~21 FLOP/byte from L2. OOB rows are clamped for loads (never stored),
// so the K-loop is EXEC-uniform: no per-iteration divergence.
// ---------------------------------------------------------------------------

#define Nn   10000
#define Dn   768
#define Hh   4
#define Ll   4
#define Rr   3
#define Ee   48000
#define D2   (Hh * Dn)   // 3072
#define Pp   512

typedef __attribute__((ext_vector_type(16))) __bf16 v16bf;
typedef __attribute__((ext_vector_type(8)))  float  v8f;

struct __align__(16) U4 { unsigned int x, y, z, w; };
union Frag { v16bf v; U4 q[2]; };

__device__ __forceinline__ unsigned short f32_to_bf16(float f) {
  unsigned int u = __float_as_uint(f);
  u += 0x7FFFu + ((u >> 16) & 1u);          // round-to-nearest-even
  return (unsigned short)(u >> 16);
}

__device__ __forceinline__ void atomicMaxF(float* addr, float v) {
  if (v >= 0.0f) atomicMax((int*)addr, __float_as_int(v));
  else           atomicMin((unsigned int*)addr, __float_as_uint(v));
}

__device__ __forceinline__ float lrelu02(float x) { return x > 0.0f ? x : 0.2f * x; }

// ---------------------------------------------------------------------------
// GEMM: C[M x Nc] = A_bf16[M x K] * Bt_bf16[Nc x K]^T (+bias, relu)
// outF (f32) and/or outB (bf16) may be null. K % 32 == 0, Nc % 64 == 0.
// Wave tile: 32 (M) x 64 (N).
// ---------------------------------------------------------------------------
__launch_bounds__(256)
__global__ void gemm_bf16_kernel(const unsigned short* __restrict__ A,
                                 const unsigned short* __restrict__ Bt,
                                 const float* __restrict__ bias,
                                 float* __restrict__ outF,
                                 unsigned short* __restrict__ outB,
                                 int M, int Nc, int K, int relu) {
  int wave = blockIdx.x * (blockDim.x >> 5) + (threadIdx.x >> 5);
  int lane = threadIdx.x & 31;
  int tilesN = Nc >> 6;                         // 64-wide macro tiles
  int tilesM = (M + 31) >> 5;                   // 32-tall macro tiles
  if (wave >= tilesM * tilesN) return;          // whole-wave uniform exit
  int tm = wave / tilesN;
  int tn = wave - tm * tilesN;
  int half = lane >> 4;
  int l16  = lane & 15;

  // A fragment rows for the two M-subtiles (clamped: OOB rows load row M-1,
  // results for those rows are computed but never stored).
  int row0 = tm * 32 + l16;
  int row1 = row0 + 16;
  int r0c = row0 < M ? row0 : (M - 1);
  int r1c = row1 < M ? row1 : (M - 1);
  int colBase = tn * 64;

  const unsigned short* a0 = A + (size_t)r0c * K + half * 8;            // 16B aligned
  const unsigned short* a1 = A + (size_t)r1c * K + half * 8;
  const unsigned short* b0 = Bt + (size_t)(colBase + l16) * K + half * 8;  // 16B aligned
  const size_t bstep = (size_t)16 * K;          // next 16-column group of Bt

  v8f acc[8];
  #pragma unroll
  for (int i = 0; i < 8; ++i) acc[i] = (v8f){0.f,0.f,0.f,0.f,0.f,0.f,0.f,0.f};

  for (int k0 = 0; k0 < K; k0 += 32) {
    Frag fa0, fa1, fb0, fb1, fb2, fb3;
    fa0.q[0] = *(const U4*)(a0 + k0);           // K = half*8    .. +7
    fa0.q[1] = *(const U4*)(a0 + k0 + 16);      // K = 16+half*8 .. +7
    fa1.q[0] = *(const U4*)(a1 + k0);
    fa1.q[1] = *(const U4*)(a1 + k0 + 16);
    fb0.q[0] = *(const U4*)(b0 + k0);           // K = half*8    .. +7
    fb0.q[1] = *(const U4*)(b0 + k0 + 16);      // K = 16+half*8 .. +7
    fb1.q[0] = *(const U4*)(b0 + bstep + k0);
    fb1.q[1] = *(const U4*)(b0 + bstep + k0 + 16);
    fb2.q[0] = *(const U4*)(b0 + 2 * bstep + k0);
    fb2.q[1] = *(const U4*)(b0 + 2 * bstep + k0 + 16);
    fb3.q[0] = *(const U4*)(b0 + 3 * bstep + k0);
    fb3.q[1] = *(const U4*)(b0 + 3 * bstep + k0 + 16);

    // warm L2->WGP path for the next K-step (global_prefetch_b8)
    if (k0 + 32 < K) {
      __builtin_prefetch(a0 + k0 + 32, 0, 0);
      __builtin_prefetch(a1 + k0 + 32, 0, 0);
      __builtin_prefetch(b0 + k0 + 32, 0, 0);
      __builtin_prefetch(b0 + 3 * bstep + k0 + 32, 0, 0);
    }

    acc[0] = __builtin_amdgcn_wmma_f32_16x16x32_bf16(false, fa0.v, false, fb0.v, (short)0, acc[0], false, false);
    acc[1] = __builtin_amdgcn_wmma_f32_16x16x32_bf16(false, fa0.v, false, fb1.v, (short)0, acc[1], false, false);
    acc[2] = __builtin_amdgcn_wmma_f32_16x16x32_bf16(false, fa0.v, false, fb2.v, (short)0, acc[2], false, false);
    acc[3] = __builtin_amdgcn_wmma_f32_16x16x32_bf16(false, fa0.v, false, fb3.v, (short)0, acc[3], false, false);
    acc[4] = __builtin_amdgcn_wmma_f32_16x16x32_bf16(false, fa1.v, false, fb0.v, (short)0, acc[4], false, false);
    acc[5] = __builtin_amdgcn_wmma_f32_16x16x32_bf16(false, fa1.v, false, fb1.v, (short)0, acc[5], false, false);
    acc[6] = __builtin_amdgcn_wmma_f32_16x16x32_bf16(false, fa1.v, false, fb2.v, (short)0, acc[6], false, false);
    acc[7] = __builtin_amdgcn_wmma_f32_16x16x32_bf16(false, fa1.v, false, fb3.v, (short)0, acc[7], false, false);
  }

  // Epilogue: acc[i*4+j] is 16x16 tile (M-subtile i, N-subtile j).
  // C layout: VGPR v -> row v + 8*half, lane%16 -> col.
  #pragma unroll
  for (int i = 0; i < 2; ++i) {
    #pragma unroll
    for (int j = 0; j < 4; ++j) {
      int c = colBase + j * 16 + l16;
      float bcol = bias ? bias[c] : 0.0f;
      #pragma unroll
      for (int v = 0; v < 8; ++v) {
        int r = tm * 32 + i * 16 + v + 8 * half;
        if (r < M) {
          float val = acc[i * 4 + j][v] + bcol;
          if (relu) val = fmaxf(val, 0.0f);
          size_t idx = (size_t)r * Nc + c;
          if (outF) outF[idx] = val;
          if (outB) outB[idx] = f32_to_bf16(val);
        }
      }
    }
  }
}

// ---------------------------------------------------------------------------
// Helpers / graph kernels
// ---------------------------------------------------------------------------
__global__ void transpose_bf16_kernel(const float* __restrict__ src,
                                      unsigned short* __restrict__ dst,
                                      int rows, int cols) {
  int idx = blockIdx.x * blockDim.x + threadIdx.x;
  if (idx >= rows * cols) return;
  int r = idx / cols, c = idx - r * cols;
  dst[(size_t)c * rows + r] = f32_to_bf16(src[idx]);
}

__global__ void init_h_kernel(const float* __restrict__ nf,
                              float* __restrict__ h, float* __restrict__ h_new,
                              unsigned short* __restrict__ hb, int n) {
  int i = blockIdx.x * blockDim.x + threadIdx.x;
  if (i >= n) return;
  float v = nf[i];
  h[i] = v; h_new[i] = 0.0f; hb[i] = f32_to_bf16(v);
}

// one wave per (node, head): alpha_s/d = <g[n,h,:], att_src/dst[h,:]>
__global__ void alpha_kernel(const float* __restrict__ g,
                             const float* __restrict__ att_src,
                             const float* __restrict__ att_dst,
                             float* __restrict__ as, float* __restrict__ ad) {
  int wid  = blockIdx.x * (blockDim.x >> 5) + (threadIdx.x >> 5);
  int lane = threadIdx.x & 31;
  if (wid >= Nn * Hh) return;
  int n = wid / Hh, h = wid - n * Hh;
  const float* gp = g + (size_t)n * D2 + h * Dn;
  const float* s  = att_src + h * Dn;
  const float* d  = att_dst + h * Dn;
  float ss = 0.f, sd = 0.f;
  for (int k = lane; k < Dn; k += 32) {
    float gv = gp[k];
    ss += gv * s[k];
    sd += gv * d[k];
  }
  for (int off = 16; off > 0; off >>= 1) {
    ss += __shfl_down(ss, off, 32);
    sd += __shfl_down(sd, off, 32);
  }
  if (lane == 0) { as[n * Hh + h] = ss; ad[n * Hh + h] = sd; }
}

__global__ void node_init_kernel(const float* __restrict__ as,
                                 const float* __restrict__ ad,
                                 float* __restrict__ eself, float* __restrict__ m,
                                 int count) {
  int i = blockIdx.x * blockDim.x + threadIdx.x;
  if (i >= count) return;
  float e = lrelu02(as[i] + ad[i]);
  eself[i] = e; m[i] = e;                       // self-loop seeds the max
}

__global__ void edge_max_kernel(const int* __restrict__ src, const int* __restrict__ dst,
                                const int* __restrict__ et, int r,
                                const float* __restrict__ as, const float* __restrict__ ad,
                                float* __restrict__ m) {
  int e = blockIdx.x * blockDim.x + threadIdx.x;
  if (e >= Ee || et[e] != r) return;
  int s = src[e], d = dst[e];
  #pragma unroll
  for (int h = 0; h < Hh; ++h)
    atomicMaxF(&m[d * Hh + h], lrelu02(as[s * Hh + h] + ad[d * Hh + h]));
}

__global__ void node_denom_kernel(const float* __restrict__ eself,
                                  const float* __restrict__ m,
                                  float* __restrict__ denom, int count) {
  int i = blockIdx.x * blockDim.x + threadIdx.x;
  if (i >= count) return;
  denom[i] = __expf(eself[i] - m[i]);           // p_self seeds the sum
}

__global__ void edge_p_kernel(const int* __restrict__ src, const int* __restrict__ dst,
                              const int* __restrict__ et, int r,
                              const float* __restrict__ as, const float* __restrict__ ad,
                              const float* __restrict__ m,
                              float* __restrict__ p, float* __restrict__ denom) {
  int e = blockIdx.x * blockDim.x + threadIdx.x;
  if (e >= Ee) return;
  if (et[e] != r) {
    #pragma unroll
    for (int h = 0; h < Hh; ++h) p[(size_t)e * Hh + h] = 0.0f;
    return;
  }
  int s = src[e], d = dst[e];
  #pragma unroll
  for (int h = 0; h < Hh; ++h) {
    float v  = lrelu02(as[s * Hh + h] + ad[d * Hh + h]);
    float pe = __expf(v - m[d * Hh + h]);
    p[(size_t)e * Hh + h] = pe;
    atomicAdd(&denom[d * Hh + h], pe);
  }
}

// one block per edge: h_new[dst,d] += (1/H) * sum_h p/denom * g[src,h,d]
__global__ void edge_msg_kernel(const int* __restrict__ src, const int* __restrict__ dst,
                                const int* __restrict__ et, int r,
                                const float* __restrict__ p, const float* __restrict__ denom,
                                const float* __restrict__ g, float* __restrict__ h_new) {
  int e = blockIdx.x;
  if (et[e] != r) return;                       // uniform across block
  int s = src[e], d = dst[e];
  __shared__ float c[Hh];
  if (threadIdx.x < Hh)
    c[threadIdx.x] = p[(size_t)e * Hh + threadIdx.x] /
                     denom[(size_t)d * Hh + threadIdx.x] * (1.0f / Hh);
  __syncthreads();
  const float* gs = g + (size_t)s * D2;
  float* hd = h_new + (size_t)d * Dn;
  for (int j = threadIdx.x; j < D2; j += blockDim.x) {
    int h = j / Dn;
    atomicAdd(&hd[j - h * Dn], c[h] * gs[j]);
  }
}

// self-loop contribution + gat_bias (bias added once per relation, as in ref)
__global__ void node_self_bias_kernel(const float* __restrict__ eself,
                                      const float* __restrict__ m,
                                      const float* __restrict__ denom,
                                      const float* __restrict__ g,
                                      const float* __restrict__ bias_l,
                                      float* __restrict__ h_new) {
  int idx = blockIdx.x * blockDim.x + threadIdx.x;
  if (idx >= Nn * Dn) return;
  int n = idx / Dn, d = idx - n * Dn;
  float acc = bias_l[d];
  #pragma unroll
  for (int h = 0; h < Hh; ++h) {
    float ps = __expf(eself[n * Hh + h] - m[n * Hh + h]);
    acc += ps / denom[n * Hh + h] * (1.0f / Hh) * g[(size_t)n * D2 + h * Dn + d];
  }
  h_new[idx] += acc;
}

__global__ void residual_relu_kernel(float* __restrict__ h, float* __restrict__ h_new,
                                     unsigned short* __restrict__ hb, int n) {
  int i = blockIdx.x * blockDim.x + threadIdx.x;
  if (i >= n) return;
  float v = fmaxf(h[i] + h_new[i], 0.0f);
  h[i] = v; h_new[i] = 0.0f; hb[i] = f32_to_bf16(v);
}

__global__ void copy_f32_kernel(const float* __restrict__ srcp, float* __restrict__ dstp, int n) {
  int i = blockIdx.x * blockDim.x + threadIdx.x;
  if (i < n) dstp[i] = srcp[i];
}

// ---------------------------------------------------------------------------
// Host orchestration
// ---------------------------------------------------------------------------
static inline void launch_gemm(const unsigned short* A, const unsigned short* Bt,
                               const float* bias, float* outF, unsigned short* outB,
                               int M, int Nc, int K, int relu, hipStream_t s) {
  int waves  = ((M + 31) / 32) * (Nc / 64);
  int blocks = (waves + 7) / 8;                 // 256 threads = 8 waves
  hipLaunchKernelGGL(gemm_bf16_kernel, dim3(blocks), dim3(256), 0, s,
                     A, Bt, bias, outF, outB, M, Nc, K, relu);
}

extern "C" void kernel_launch(void* const* d_in, const int* in_sizes, int n_in,
                              void* d_out, int out_size, void* d_ws, size_t ws_size,
                              hipStream_t stream) {
  const float* node_features = (const float*)d_in[0];
  const int*   edge_index    = (const int*)  d_in[1];
  const int*   edge_type     = (const int*)  d_in[2];
  const float* rel_W         = (const float*)d_in[3];
  const float* rel_b         = (const float*)d_in[4];
  const float* gat_W         = (const float*)d_in[5];
  const float* gat_att_src   = (const float*)d_in[6];
  const float* gat_att_dst   = (const float*)d_in[7];
  const float* gat_bias      = (const float*)d_in[8];
  const float* proj_W1       = (const float*)d_in[9];
  const float* proj_b1       = (const float*)d_in[10];
  const float* proj_W2       = (const float*)d_in[11];
  const float* proj_b2       = (const float*)d_in[12];
  const int* srcI = edge_index;
  const int* dstI = edge_index + Ee;

  // bump allocator over d_ws (total ~251 MB)
  char* wsb = (char*)d_ws;
  size_t off = 0;
  auto alloc = [&](size_t bytes) -> void* {
    void* pp = wsb + off;
    off += (bytes + 255) & ~(size_t)255;
    return pp;
  };
  float*          h     = (float*)alloc((size_t)Nn * Dn * 4);
  float*          h_new = (float*)alloc((size_t)Nn * Dn * 4);
  unsigned short* h_bf  = (unsigned short*)alloc((size_t)Nn * Dn * 2);
  unsigned short* t_bf  = (unsigned short*)alloc((size_t)Nn * Dn * 2);
  unsigned short* z_bf  = (unsigned short*)alloc((size_t)Nn * Pp * 2);
  float*          g     = (float*)alloc((size_t)Nn * D2 * 4);
  float*          as    = (float*)alloc((size_t)Nn * Hh * 4);
  float*          ad    = (float*)alloc((size_t)Nn * Hh * 4);
  float*          m     = (float*)alloc((size_t)Nn * Hh * 4);
  float*          denom = (float*)alloc((size_t)Nn * Hh * 4);
  float*          eself = (float*)alloc((size_t)Nn * Hh * 4);
  float*          pbuf  = (float*)alloc((size_t)Ee * Hh * 4);
  unsigned short* relWT = (unsigned short*)alloc((size_t)Rr * Dn * Dn * 2);
  unsigned short* gatWT = (unsigned short*)alloc((size_t)Ll * D2 * Dn * 2);
  unsigned short* W1T   = (unsigned short*)alloc((size_t)Pp * Dn * 2);
  unsigned short* W2T   = (unsigned short*)alloc((size_t)Dn * Pp * 2);

  const int TB = 256;
  auto blks = [](size_t n) { return (unsigned)((n + 255) / 256); };

  // --- weight prep: transpose + convert to bf16 -------------------------
  for (int r = 0; r < Rr; ++r)
    hipLaunchKernelGGL(transpose_bf16_kernel, dim3(blks((size_t)Dn * Dn)), dim3(TB), 0, stream,
                       rel_W + (size_t)r * Dn * Dn, relWT + (size_t)r * Dn * Dn, Dn, Dn);
  for (int l = 0; l < Ll; ++l)
    hipLaunchKernelGGL(transpose_bf16_kernel, dim3(blks((size_t)Dn * D2)), dim3(TB), 0, stream,
                       gat_W + (size_t)l * Dn * D2, gatWT + (size_t)l * D2 * Dn, Dn, D2);
  hipLaunchKernelGGL(transpose_bf16_kernel, dim3(blks((size_t)Dn * Pp)), dim3(TB), 0, stream,
                     proj_W1, W1T, Dn, Pp);
  hipLaunchKernelGGL(transpose_bf16_kernel, dim3(blks((size_t)Pp * Dn)), dim3(TB), 0, stream,
                     proj_W2, W2T, Pp, Dn);

  // --- init -------------------------------------------------------------
  hipLaunchKernelGGL(init_h_kernel, dim3(blks((size_t)Nn * Dn)), dim3(TB), 0, stream,
                     node_features, h, h_new, h_bf, Nn * Dn);

  // --- layers -----------------------------------------------------------
  for (int l = 0; l < Ll; ++l) {
    for (int r = 0; r < Rr; ++r) {
      // t = h @ rel_W[r] + rel_b[r]   (bf16 out)
      launch_gemm(h_bf, relWT + (size_t)r * Dn * Dn, rel_b + (size_t)r * Dn,
                  nullptr, t_bf, Nn, Dn, Dn, 0, stream);
      // g = t @ gat_W[l]              (f32 out, [N, H*D])
      launch_gemm(t_bf, gatWT + (size_t)l * D2 * Dn, nullptr,
                  g, nullptr, Nn, D2, Dn, 0, stream);
      // attention logits per (node, head)
      hipLaunchKernelGGL(alpha_kernel, dim3((Nn * Hh + 7) / 8), dim3(TB), 0, stream,
                         g, gat_att_src + (size_t)l * Hh * Dn,
                         gat_att_dst + (size_t)l * Hh * Dn, as, ad);
      hipLaunchKernelGGL(node_init_kernel, dim3(blks(Nn * Hh)), dim3(TB), 0, stream,
                         as, ad, eself, m, Nn * Hh);
      hipLaunchKernelGGL(edge_max_kernel, dim3(blks(Ee)), dim3(TB), 0, stream,
                         srcI, dstI, edge_type, r, as, ad, m);
      hipLaunchKernelGGL(node_denom_kernel, dim3(blks(Nn * Hh)), dim3(TB), 0, stream,
                         eself, m, denom, Nn * Hh);
      hipLaunchKernelGGL(edge_p_kernel, dim3(blks(Ee)), dim3(TB), 0, stream,
                         srcI, dstI, edge_type, r, as, ad, m, pbuf, denom);
      hipLaunchKernelGGL(edge_msg_kernel, dim3(Ee), dim3(TB), 0, stream,
                         srcI, dstI, edge_type, r, pbuf, denom, g, h_new);
      hipLaunchKernelGGL(node_self_bias_kernel, dim3(blks((size_t)Nn * Dn)), dim3(TB), 0, stream,
                         eself, m, denom, g, gat_bias + (size_t)l * Dn, h_new);
    }
    // h = relu(h + h_new); h_new = 0; refresh bf16 copy
    hipLaunchKernelGGL(residual_relu_kernel, dim3(blks((size_t)Nn * Dn)), dim3(TB), 0, stream,
                       h, h_new, h_bf, Nn * Dn);
  }

  // --- output head ------------------------------------------------------
  float* positions = (float*)d_out;             // [N, Dn]
  float* h_out     = positions + (size_t)Nn * Dn;
  // z = relu(h @ W1 + b1) -> bf16
  launch_gemm(h_bf, W1T, proj_b1, nullptr, z_bf, Nn, Pp, Dn, 1, stream);
  // positions = z @ W2 + b2 -> f32 straight into d_out
  launch_gemm(z_bf, W2T, proj_b2, positions, nullptr, Nn, Dn, Pp, 0, stream);
  // second output: final h
  hipLaunchKernelGGL(copy_f32_kernel, dim3(blks((size_t)Nn * Dn)), dim3(TB), 0, stream,
                     h, h_out, Nn * Dn);
}
